// HROMAttention_27341761806559
// MI455X (gfx1250) — matrix operand order repaired and checked by reading, not compiled, not run on hardware
//
#include <hip/hip_runtime.h>
#include <hip/hip_bf16.h>

// ---------------- problem constants ----------------
#define BATCH   4
#define SEQ     2048
#define DMODEL  768
#define NHEADS  8
#define HDIM    96          // 6 x 16
#define QKVN    (3*DMODEL)  // 2304
#define NTOK    (BATCH*SEQ) // 8192
#define KDIM    768         // contraction dim for both GEMMs

typedef __attribute__((ext_vector_type(16))) __bf16 v16bf;
typedef __attribute__((ext_vector_type(8)))  float  v8f;

union FragBF { v16bf v; uint4 q[2]; };

// WMMA bf16: D = A(16x32) * B(32x16) + C, f32 accumulate
__device__ __forceinline__ v8f wmma_bf16(v16bf a, v16bf b, v8f c) {
    return __builtin_amdgcn_wmma_f32_16x16x32_bf16(
        false, a, false, b, (short)0, c, false, false);
}

// A-fragment (16x32 bf16, row-major source, leading dim = lda elements).
// lanes 0-15: row m=lane, K {0..7,16..23}; lanes 16-31: row m=lane-16,
// K {8..15,24..31}. base points at element (row0, k0).
__device__ __forceinline__ v16bf load_frag_a(const __bf16* base, int lda) {
    int l = threadIdx.x & 31;
    int m = l & 15, h = l >> 4;
    const __bf16* p = base + m * lda + h * 8;
    FragBF f;
    f.q[0] = *(const uint4*)(p);
    f.q[1] = *(const uint4*)(p + 16);
    return f.v;
}

// B-fragment (32x16 bf16) from W[N][K] row-major (B[k][n] = W[n][k]).
// lane holds column n = lane&15; lanes 0-15 K {0..7,16..23}, lanes 16-31 K {8..15,24..31}.
__device__ __forceinline__ v16bf load_frag_b(const __bf16* base, int ldb) {
    int l = threadIdx.x & 31;
    int n = l & 15, h = l >> 4;
    const __bf16* p = base + n * ldb + h * 8;
    FragBF f;
    f.q[0] = *(const uint4*)(p);
    f.q[1] = *(const uint4*)(p + 16);
    return f.v;
}

// ---- CDNA5 async global->LDS copy (16B per lane) + waits ------------------
__device__ __forceinline__ void async_b128(unsigned lds_byte_off, const __bf16* g) {
    asm volatile("global_load_async_to_lds_b128 %0, %1, off"
                 :: "v"(lds_byte_off), "v"(g)
                 : "memory");
}
__device__ __forceinline__ void wait_async0() {
    asm volatile("s_wait_asynccnt 0x0" ::: "memory");
}
// allow the next chunk's 6 async ops to stay in flight (retire in order)
__device__ __forceinline__ void wait_async6() {
    asm volatile("s_wait_asynccnt 0x6" ::: "memory");
}
__device__ __forceinline__ unsigned lds_off(const void* p) {
    return (unsigned)(unsigned long long)p;
}

// ---------------- fp32 -> bf16 convert ----------------
__global__ void k_f32_to_bf16(const float* __restrict__ in,
                              __bf16* __restrict__ out, int n) {
    int i = blockIdx.x * blockDim.x + threadIdx.x;
    if (i < n) out[i] = (__bf16)in[i];
}

// ---------------- WMMA GEMM:  C[M,N] = A[M,768] * W[N,768]^T + bias --------
// block = 128 threads (4 waves); wave w computes a 16x64 strip.
// 16x768 A row-block async-staged to LDS once, shared by 4 waves.
// k-loop: 2x-unrolled ping-pong register sets (no copy-back movs); with
// KDIM compile-time and full unroll every guard is constant, so loads for
// step i+2 are issued between the WMMA bursts of steps i and i+1.
// grid = (M/16, N/256)
__launch_bounds__(128)
__global__ void k_gemm_bf16(const __bf16* __restrict__ A,
                            const __bf16* __restrict__ W,
                            const float*  __restrict__ bias,
                            float* __restrict__ C,
                            int M, int N) {
    __shared__ __bf16 Al[16 * KDIM];                 // 24KB
    const int wave = threadIdx.x >> 5;
    const int tid  = threadIdx.x;
    const int row0 = blockIdx.x * 16;
    const int col0 = (blockIdx.y * 4 + wave) * 64;

    // ---- async stage A block: 12288 bf16 = 24KB, 16B/lane x 12 iters ----
    {
        const __bf16* gA = A + (size_t)row0 * KDIM;
        unsigned base = lds_off(&Al[0]);
        #pragma unroll
        for (int it = 0; it < 12; ++it) {
            int e = tid * 8 + it * 1024;
            async_b128(base + e * 2, gA + e);
        }
    }
    wait_async0();
    __syncthreads();

    v8f zero = {};
    v8f acc[4];
    #pragma unroll
    for (int j = 0; j < 4; ++j) acc[j] = zero;

    const __bf16* Wb = W + (size_t)col0 * KDIM;

    // ping set (k = 0)
    v16bf aP = load_frag_a(&Al[0], KDIM);
    v16bf bP[4];
    #pragma unroll
    for (int j = 0; j < 4; ++j)
        bP[j] = load_frag_b(Wb + (size_t)j * 16 * KDIM, KDIM);

    #pragma unroll
    for (int k0 = 0; k0 < KDIM; k0 += 64) {
        // pong set (k0 + 32): always in range (KDIM % 64 == 0)
        __builtin_prefetch(Wb + k0 + 96, 0, 1);      // global_prefetch_b8
        v16bf aQ = load_frag_a(&Al[k0 + 32], KDIM);
        v16bf bQ[4];
        #pragma unroll
        for (int j = 0; j < 4; ++j)
            bQ[j] = load_frag_b(Wb + (size_t)j * 16 * KDIM + k0 + 32, KDIM);
        #pragma unroll
        for (int j = 0; j < 4; ++j)
            acc[j] = wmma_bf16(aP, bP[j], acc[j]);

        // refill ping set for k0 + 64 (compile-time guard)
        if (k0 + 64 < KDIM) {
            aP = load_frag_a(&Al[k0 + 64], KDIM);
            #pragma unroll
            for (int j = 0; j < 4; ++j)
                bP[j] = load_frag_b(Wb + (size_t)j * 16 * KDIM + k0 + 64, KDIM);
        }
        #pragma unroll
        for (int j = 0; j < 4; ++j)
            acc[j] = wmma_bf16(aQ, bQ[j], acc[j]);
    }

    const int l  = threadIdx.x & 31;
    const int lm = l & 15, lh = l >> 4;
    #pragma unroll
    for (int j = 0; j < 4; ++j) {
        int n = col0 + j * 16 + lm;
        float bv = bias[n];
        #pragma unroll
        for (int r = 0; r < 8; ++r) {
            int m = row0 + r + lh * 8;
            C[(size_t)m * N + n] = acc[j][r] + bv;
        }
    }
}

// ---------------- RoPE + head-major scatter ----------------
__global__ void k_rope_scatter(const float* __restrict__ qkv,
                               __bf16* __restrict__ Qr,
                               __bf16* __restrict__ Kr,
                               __bf16* __restrict__ Vt) {
    int idx = blockIdx.x * blockDim.x + threadIdx.x;
    const int total = BATCH * NHEADS * SEQ * HDIM;
    if (idx >= total) return;
    int d = idx % HDIM;
    int t = (idx / HDIM) % SEQ;
    int h = (idx / (HDIM * SEQ)) % NHEADS;
    int b = idx / (HDIM * SEQ * NHEADS);

    size_t rowbase = ((size_t)b * SEQ + t) * QKVN;
    int coff = h * HDIM + d;
    float q = qkv[rowbase + coff];
    float k = qkv[rowbase + DMODEL + coff];
    float v = qkv[rowbase + 2 * DMODEL + coff];

    int i = (d < 48) ? d : d - 48;
    float inv = __powf(10000.0f, -(2.0f * i) / 96.0f);
    float ang = (float)t * inv;
    float cs = __cosf(ang), sn = __sinf(ang);
    int dro  = (d < 48) ? d + 48 : d - 48;
    float sgn = (d < 48) ? -1.0f : 1.0f;
    float qro = qkv[rowbase + h * HDIM + dro];
    float kro = qkv[rowbase + DMODEL + h * HDIM + dro];
    float qr = q * cs + sgn * qro * sn;
    float kr = k * cs + sgn * kro * sn;

    int bh = b * NHEADS + h;
    Qr[((size_t)bh * SEQ + t) * HDIM + d] = (__bf16)qr;
    Kr[((size_t)bh * SEQ + t) * HDIM + d] = (__bf16)kr;
    Vt[((size_t)bh * HDIM + d) * SEQ + t] = (__bf16)v;
}

// ---------------- flash attention ----------------
// grid = (B*H, T/64); block = 128 (4 waves); wave owns a 16-query tile.
// Ping-pong LDS K/V buffers: chunk i+1 is async-DMA'd while chunk i is
// computed. Each wave issues exactly 6 async ops per chunk (3 K + 3 V);
// asyncs retire in order, so s_wait_asynccnt<=6 isolates chunk i.
__launch_bounds__(128)
__global__ void k_flash_attn(const __bf16* __restrict__ Qr,
                             const __bf16* __restrict__ Kr,
                             const __bf16* __restrict__ Vt,
                             __bf16* __restrict__ Oout) {
    __shared__ __bf16 Kl[2][32 * HDIM];             // 2 x 6KB  Kl[t][d]
    __shared__ __bf16 Vl[2][HDIM * 32];             // 2 x 6KB  Vl[d][t]
    __shared__ __bf16 pstage[4 * 16 * 32];          // 4KB, 1KB per wave

    const int wave = threadIdx.x >> 5;
    const int tid  = threadIdx.x;
    const int l  = tid & 31;
    const int lm = l & 15, lh = l >> 4;

    const int bh = blockIdx.x;
    const int b = bh / NHEADS, h = bh % NHEADS;
    const int qblk  = blockIdx.y * 64;
    const int qbase = qblk + wave * 16;

    const __bf16* Qp = Qr + (size_t)bh * SEQ * HDIM;
    const __bf16* Kp = Kr + (size_t)bh * SEQ * HDIM;
    const __bf16* Vp = Vt + (size_t)bh * HDIM * SEQ;

    v16bf qf[3];
    #pragma unroll
    for (int c = 0; c < 3; ++c)
        qf[c] = load_frag_a(Qp + (size_t)qbase * HDIM + c * 32, HDIM);

    v8f zero = {};
    v8f o[6];
    #pragma unroll
    for (int dt = 0; dt < 6; ++dt) o[dt] = zero;
    float mrow[8], lrow[8];
    #pragma unroll
    for (int r = 0; r < 8; ++r) { mrow[r] = -1e30f; lrow[r] = 0.0f; }

    const float scale = 0.1020620726f;               // 1/sqrt(96)
    __bf16* pbuf = &pstage[wave * 16 * 32];
    unsigned kl_base[2] = { lds_off(&Kl[0][0]), lds_off(&Kl[1][0]) };
    unsigned vl_base[2] = { lds_off(&Vl[0][0]), lds_off(&Vl[1][0]) };

    const int nchunks = qblk / 32 + 2;               // (qblk+64)/32

    // block-cooperative async stage of one 32-token K+V chunk
    auto stage = [&](int tb, int buf) {
        const __bf16* gK = Kp + (size_t)tb * HDIM;   // contiguous 3072 bf16
        #pragma unroll
        for (int it = 0; it < 3; ++it) {
            int e = tid * 8 + it * 1024;
            async_b128(kl_base[buf] + e * 2, gK + e);
        }
        int toff = (tid & 3) * 8;                    // 4 lanes of 16B per d-row
        #pragma unroll
        for (int it = 0; it < 3; ++it) {
            int d = (tid >> 2) + it * 32;
            async_b128(vl_base[buf] + (d * 32 + toff) * 2,
                       Vp + (size_t)d * SEQ + tb + toff);
        }
    };

    stage(0, 0);                                     // prologue: chunk 0
    for (int ci = 0; ci < nchunks; ++ci) {
        const int tb  = ci * 32;
        const int buf = ci & 1;
        const bool has_next = (ci + 1 < nchunks);
        if (has_next) stage(tb + 32, buf ^ 1);       // overlap next DMA
        if (has_next) wait_async6(); else wait_async0();
        __syncthreads();

        if (tb <= qbase + 15) {
            const __bf16* Kc = &Kl[buf][0];
            const __bf16* Vc = &Vl[buf][0];
            // ---- S = Q K^T for 32 keys (two 16-wide tiles) ----
            v8f s0 = zero, s1 = zero;
            #pragma unroll
            for (int c = 0; c < 3; ++c) {
                v16bf kb0 = load_frag_b(Kc + c * 32, HDIM);
                v16bf kb1 = load_frag_b(Kc + 16 * HDIM + c * 32, HDIM);
                s0 = wmma_bf16(qf[c], kb0, s0);
                s1 = wmma_bf16(qf[c], kb1, s1);
            }
            // ---- causal mask + online softmax ----
            #pragma unroll
            for (int r = 0; r < 8; ++r) {
                int m  = qbase + r + lh * 8;
                int n0 = tb + lm, n1 = n0 + 16;
                float a0 = s0[r] * scale + ((n0 <= m) ? 0.0f : -1e30f);
                float a1 = s1[r] * scale + ((n1 <= m) ? 0.0f : -1e30f);
                float mx = fmaxf(a0, a1);
                #pragma unroll
                for (int off = 8; off >= 1; off >>= 1)
                    mx = fmaxf(mx, __shfl_xor(mx, off, 32));
                float mnew = fmaxf(mrow[r], mx);
                float corr = __expf(mrow[r] - mnew);
                float p0 = __expf(a0 - mnew);
                float p1 = __expf(a1 - mnew);
                float rs = p0 + p1;
                #pragma unroll
                for (int off = 8; off >= 1; off >>= 1)
                    rs += __shfl_xor(rs, off, 32);
                lrow[r] = lrow[r] * corr + rs;
                mrow[r] = mnew;
                #pragma unroll
                for (int dt = 0; dt < 6; ++dt) o[dt][r] *= corr;
                int mm = r + lh * 8;
                pbuf[mm * 32 + lm]      = (__bf16)p0;
                pbuf[mm * 32 + lm + 16] = (__bf16)p1;
            }
            // reload P as a 16x32 A fragment (same-wave LDS, in-order)
            v16bf pf = load_frag_a(pbuf, 32);
            // ---- O += P * V over six 16-wide d tiles ----
            #pragma unroll
            for (int dt = 0; dt < 6; ++dt) {
                v16bf vf = load_frag_b(Vc + dt * 16 * 32, 32);
                o[dt] = wmma_bf16(pf, vf, o[dt]);
            }
        }
        __syncthreads();   // all waves done reading buf before it is rewritten
    }

    // ---- epilogue: normalize, write bf16 [NTOK][768] token-major ----
    #pragma unroll
    for (int dt = 0; dt < 6; ++dt) {
        #pragma unroll
        for (int r = 0; r < 8; ++r) {
            int m = qbase + r + lh * 8;
            int n = h * HDIM + dt * 16 + lm;
            Oout[((size_t)(b * SEQ + m)) * DMODEL + n] = (__bf16)(o[dt][r] / lrow[r]);
        }
    }
}

// ---------------- host launcher ----------------
extern "C" void kernel_launch(void* const* d_in, const int* in_sizes, int n_in,
                              void* d_out, int out_size, void* d_ws, size_t ws_size,
                              hipStream_t stream) {
    const float* x      = (const float*)d_in[0];
    // d_in[1] = mask (ignored; causal mask computed analytically)
    const float* w_qkv  = (const float*)d_in[2];
    const float* b_qkv  = (const float*)d_in[3];
    const float* w_proj = (const float*)d_in[4];
    const float* b_proj = (const float*)d_in[5];
    float* out = (float*)d_out;

    auto align = [](size_t v) { return (v + 255) & ~(size_t)255; };
    char* ws = (char*)d_ws;
    size_t off = 0;
    __bf16* xb     = (__bf16*)(ws + off); off += align((size_t)NTOK * DMODEL * 2);
    __bf16* wqkvb  = (__bf16*)(ws + off); off += align((size_t)QKVN * DMODEL * 2);
    __bf16* wprojb = (__bf16*)(ws + off); off += align((size_t)DMODEL * DMODEL * 2);
    float*  qkvf   = (float*) (ws + off); off += align((size_t)NTOK * QKVN * 4);
    __bf16* Qr     = (__bf16*)(ws + off); off += align((size_t)NTOK * DMODEL * 2);
    __bf16* Kr     = (__bf16*)(ws + off); off += align((size_t)NTOK * DMODEL * 2);
    __bf16* Vt     = (__bf16*)(ws + off); off += align((size_t)NTOK * DMODEL * 2);
    __bf16* attn   = (__bf16*)(ws + off); off += align((size_t)NTOK * DMODEL * 2);
    (void)ws_size; (void)in_sizes; (void)n_in; (void)out_size;

    {
        int n1 = NTOK * DMODEL;
        k_f32_to_bf16<<<(n1 + 255) / 256, 256, 0, stream>>>(x, xb, n1);
        int n2 = QKVN * DMODEL;
        k_f32_to_bf16<<<(n2 + 255) / 256, 256, 0, stream>>>(w_qkv, wqkvb, n2);
        int n3 = DMODEL * DMODEL;
        k_f32_to_bf16<<<(n3 + 255) / 256, 256, 0, stream>>>(w_proj, wprojb, n3);
    }
    {
        dim3 grid(NTOK / 16, QKVN / 256);
        k_gemm_bf16<<<grid, 128, 0, stream>>>(xb, wqkvb, b_qkv, qkvf, NTOK, QKVN);
    }
    {
        int n = BATCH * NHEADS * SEQ * HDIM;
        k_rope_scatter<<<(n + 255) / 256, 256, 0, stream>>>(qkvf, Qr, Kr, Vt);
    }
    {
        dim3 grid(BATCH * NHEADS, SEQ / 64);
        k_flash_attn<<<grid, 128, 0, stream>>>(Qr, Kr, Vt, attn);
    }
    {
        dim3 grid(NTOK / 16, DMODEL / 256);
        k_gemm_bf16<<<grid, 128, 0, stream>>>(attn, wprojb, b_proj, out, NTOK, DMODEL);
    }
}
